// Attention_74242804679195
// MI455X (gfx1250) — matrix operand order repaired and checked by reading, not compiled, not run on hardware
//
#include <hip/hip_runtime.h>
#include <math.h>

typedef __attribute__((ext_vector_type(16))) _Float16 v16h;
typedef __attribute__((ext_vector_type(8)))  float    v8f;
typedef __attribute__((ext_vector_type(4)))  float    f4;

constexpr int B = 32, S = 4096, H = 512;
constexpr int TILE_M = 16;            // rows per wave (WMMA M)
constexpr int KSTEPS = H / 32;        // 16 WMMA k-steps of 32
constexpr int NTILES = H / 16;        // 32 output tiles of 16
constexpr int NPAIR  = NTILES / 2;    // 16 outer iterations, 2 tiles each
constexpr int WAVES  = 8;             // waves per block (256 threads, 128 rows)
constexpr int FRAG_HALVES = 512;      // one B fragment = 32 lanes * 16 halves
constexpr int NT_HALVES   = KSTEPS * FRAG_HALVES;  // 8192 halves = 16KB per nt

// ---- CDNA5 async global->LDS copy (ASYNCcnt-tracked, no VGPR round-trip) ----
__device__ __forceinline__ void async_ld_b128(void* lds, const void* gptr) {
  uint32_t loff = (uint32_t)(uintptr_t)lds;          // LDS byte offset (addr[31:0])
  uint64_t ga   = (uint64_t)(uintptr_t)gptr;
  asm volatile("global_load_async_to_lds_b128 %0, %1, off"
               :: "v"(loff), "v"(ga) : "memory");
}
__device__ __forceinline__ void wait_asynccnt0() {
  asm volatile("s_wait_asynccnt 0x0" ::: "memory");
}

// Pack two runs of 8 consecutive f32 into a v16h (A operand fragment).
// Non-temporal: enc is 256MB read-once streaming; keep w1p resident in L2.
__device__ __forceinline__ v16h pack2x8_nt(const float* p0, const float* p1) {
  const f4 a0 = __builtin_nontemporal_load((const f4*)p0);
  const f4 a1 = __builtin_nontemporal_load((const f4*)p0 + 1);
  const f4 b0 = __builtin_nontemporal_load((const f4*)p1);
  const f4 b1 = __builtin_nontemporal_load((const f4*)p1 + 1);
  v16h r;
  r[0] = (_Float16)a0.x;  r[1] = (_Float16)a0.y;  r[2] = (_Float16)a0.z;  r[3] = (_Float16)a0.w;
  r[4] = (_Float16)a1.x;  r[5] = (_Float16)a1.y;  r[6] = (_Float16)a1.z;  r[7] = (_Float16)a1.w;
  r[8] = (_Float16)b0.x;  r[9] = (_Float16)b0.y;  r[10] = (_Float16)b0.z; r[11] = (_Float16)b0.w;
  r[12] = (_Float16)b1.x; r[13] = (_Float16)b1.y; r[14] = (_Float16)b1.z; r[15] = (_Float16)b1.w;
  return r;
}

// Kernel 0: one-time w1 f32 -> f16 repack into B-fragment-major layout:
//   w1p[ ((nt*KSTEPS + k)*32 + lane) * 16 + j ]
__global__ __launch_bounds__(256) void pack_w1(const float* __restrict__ w1,
                                               _Float16* __restrict__ w1p) {
  int t    = blockIdx.x * 256 + threadIdx.x;  // 0 .. 16383
  int frag = t >> 5;                          // nt*KSTEPS + k
  int lane = t & 31;
  int m  = lane & 15, hi = lane >> 4;
  int nt = frag >> 4, k = frag & 15;
  const float* src = w1 + (size_t)(nt * 16 + m) * H + k * 32 + hi * 16;
  _Float16* dst = w1p + ((size_t)frag * 32 + lane) * 16;
  #pragma unroll
  for (int j = 0; j < 16; ++j) dst[j] = (_Float16)src[j];
}

// Kernel 1: decT[b*H + o] = decoder_state[b,:] . w2[o,:]   (tiny)
__global__ __launch_bounds__(256) void dec_proj(const float* __restrict__ dec,
                                                const float* __restrict__ w2,
                                                float* __restrict__ decT) {
  int t = blockIdx.x * blockDim.x + threadIdx.x;   // 0 .. B*H-1
  int b = t >> 9;
  int o = t & (H - 1);
  const float* x = dec + (size_t)b * H;
  const float* w = w2 + (size_t)o * H;
  float s = 0.f;
  #pragma unroll 4
  for (int h = 0; h < H; h += 4) {
    float4 xv = *(const float4*)(x + h);
    float4 wv = *(const float4*)(w + h);
    s += xv.x * wv.x + xv.y * wv.y + xv.z * wv.z + xv.w * wv.w;
  }
  decT[t] = s;
}

// Kernel 2: 8-wave block, 128 rows. A tiles resident in VGPRs (f16).
// Per outer step, TWO w1 fragment sets (nt, nt+16) are async-DMA staged into a
// 32KB half of a ping-pong LDS buffer; each wave runs two independent WMMA
// accumulation chains against the shared A registers.
__global__ __launch_bounds__(WAVES * 32)
void attn_logits(const float* __restrict__ enc, const _Float16* __restrict__ w1p,
                 const float* __restrict__ decT, const float* __restrict__ vt,
                 float* __restrict__ logits) {
  __shared__ _Float16 sbuf[2][2 * NT_HALVES];   // 2 x 32KB ping-pong

  const int lane = threadIdx.x & 31;
  const int wave = threadIdx.x >> 5;
  const int tile = blockIdx.x * WAVES + wave;       // 0 .. B*S/16-1
  const int b    = tile / (S / TILE_M);
  const int s0   = (tile % (S / TILE_M)) * TILE_M;

  const int m    = lane & 15;
  const int hi   = lane >> 4;
  const int asel = hi * 16;            // A operand K offset (contiguous: k = 16*hi + s)

  // ---- Load & convert this wave's 16x512 A tile (f32 -> f16) into VGPRs ----
  const float* arow = enc + ((size_t)b * S + s0 + m) * H;
  v16h A[KSTEPS];
  #pragma unroll
  for (int k = 0; k < KSTEPS; ++k) {
    const float* p = arow + k * 32 + asel;
    A[k] = pack2x8_nt(p, p + 8);
  }

  float accL[8];
  #pragma unroll
  for (int r = 0; r < 8; ++r) accL[r] = 0.f;

  const float* decb = decT + (size_t)b * H;

  // Async-stage fragment sets for tiles (it, it+NPAIR) into buffer bi (32KB).
  auto stage = [&](int it, int bi) {
    const _Float16* src0 = w1p + (size_t)it * NT_HALVES;
    const _Float16* src1 = w1p + (size_t)(it + NPAIR) * NT_HALVES;
    _Float16* d0 = sbuf[bi];
    _Float16* d1 = sbuf[bi] + NT_HALVES;
    #pragma unroll
    for (int i = 0; i < 4; ++i) {
      int idx = threadIdx.x + i * 256;              // 16B chunk index
      async_ld_b128(&d0[idx * 8], src0 + idx * 8);
      async_ld_b128(&d1[idx * 8], src1 + idx * 8);
    }
  };

  stage(0, 0);
  wait_asynccnt0();
  __syncthreads();

  #pragma unroll 1
  for (int it = 0; it < NPAIR; ++it) {
    if (it + 1 < NPAIR) stage(it + 1, (it + 1) & 1);

    // Two independent WMMA chains (tiles it and it+16) over K=512.
    const _Float16* fb0 = sbuf[it & 1] + lane * 16;
    const _Float16* fb1 = fb0 + NT_HALVES;
    v8f acc0 = {}, acc1 = {};
    #pragma unroll
    for (int k = 0; k < KSTEPS; ++k) {
      v16h b0 = *(const v16h*)(fb0 + k * FRAG_HALVES);
      v16h b1 = *(const v16h*)(fb1 + k * FRAG_HALVES);
      acc0 = __builtin_amdgcn_wmma_f32_16x16x32_f16(
          false, A[k], false, b0, (short)0, acc0, false, false);
      acc1 = __builtin_amdgcn_wmma_f32_16x16x32_f16(
          false, A[k], false, b1, (short)0, acc1, false, false);
    }

    // Epilogues: C/D layout -> N = lane&15 = m; M = r (lanes 0-15) / 8+r (16-31)
    const int n0 = it * 16 + m;
    const int n1 = (it + NPAIR) * 16 + m;
    const float dv0 = decb[n0], vv0 = vt[n0];
    const float dv1 = decb[n1], vv1 = vt[n1];
    #pragma unroll
    for (int r = 0; r < 8; ++r)
      accL[r] += tanhf(acc0[r] + dv0) * vv0 + tanhf(acc1[r] + dv1) * vv1;

    wait_asynccnt0();   // DMA for it+1 landed in LDS
    __syncthreads();    // all waves done with buffer it before it is re-staged
  }

  // Reduce over N across each 16-lane half (masks <16 stay in-half on wave32).
  #pragma unroll
  for (int r = 0; r < 8; ++r) {
    float v = accL[r];
    v += __shfl_xor(v, 1, 32);
    v += __shfl_xor(v, 2, 32);
    v += __shfl_xor(v, 4, 32);
    v += __shfl_xor(v, 8, 32);
    accL[r] = v;
  }
  if (lane == 0) {
    #pragma unroll
    for (int r = 0; r < 8; ++r)
      logits[(size_t)b * S + s0 + r] = accL[r];        // M = r
  } else if (lane == 16) {
    #pragma unroll
    for (int r = 0; r < 8; ++r)
      logits[(size_t)b * S + s0 + 8 + r] = accL[r];    // M = 8 + r
  }
}

// Kernel 3: softmax over S per batch row.
__global__ __launch_bounds__(256) void softmax_rows(const float* __restrict__ logits,
                                                    float* __restrict__ out) {
  const int b = blockIdx.x;
  const int t = threadIdx.x;
  const float* row = logits + (size_t)b * S;
  float vals[S / 256];
  float mx = -INFINITY;
  #pragma unroll
  for (int i = 0; i < S / 256; ++i) {
    vals[i] = row[t + i * 256];
    mx = fmaxf(mx, vals[i]);
  }
  __shared__ float red[256];
  red[t] = mx;
  __syncthreads();
  for (int off = 128; off > 0; off >>= 1) {
    if (t < off) red[t] = fmaxf(red[t], red[t + off]);
    __syncthreads();
  }
  mx = red[0];
  __syncthreads();
  float sum = 0.f;
  #pragma unroll
  for (int i = 0; i < S / 256; ++i) {
    vals[i] = __expf(vals[i] - mx);
    sum += vals[i];
  }
  red[t] = sum;
  __syncthreads();
  for (int off = 128; off > 0; off >>= 1) {
    if (t < off) red[t] += red[t + off];
    __syncthreads();
  }
  const float inv = 1.0f / red[0];
  #pragma unroll
  for (int i = 0; i < S / 256; ++i)
    out[(size_t)b * S + t + i * 256] = vals[i] * inv;
}

extern "C" void kernel_launch(void* const* d_in, const int* in_sizes, int n_in,
                              void* d_out, int out_size, void* d_ws, size_t ws_size,
                              hipStream_t stream) {
  const float* dec = (const float*)d_in[0];   // (B,H)
  const float* enc = (const float*)d_in[1];   // (B,S,H)
  const float* w1  = (const float*)d_in[2];   // (H,H)
  const float* w2  = (const float*)d_in[3];   // (H,H)
  const float* vt  = (const float*)d_in[4];   // (H,)
  float* out = (float*)d_out;                 // (B,S)

  float* decT      = (float*)d_ws;                    // B*H floats
  float* logits    = decT + (size_t)B * H;            // B*S floats
  _Float16* w1p    = (_Float16*)(logits + (size_t)B * S);  // H*H halves (512KB)

  pack_w1<<<dim3((H * H / 16) / 256), 256, 0, stream>>>(w1, w1p);
  dec_proj<<<dim3((B * H) / 256), 256, 0, stream>>>(dec, w2, decT);

  const int blocks = (B * S / TILE_M) / WAVES;        // 1024 blocks, 8 waves each
  attn_logits<<<dim3(blocks), WAVES * 32, 0, stream>>>(enc, w1p, decT, vt, logits);

  softmax_rows<<<dim3(B), 256, 0, stream>>>(logits, out);
}
